// EnergyPredictor_56908316672258
// MI455X (gfx1250) — compile-verified, matrix operand order checked by reading, not run on hardware
//
#include <hip/hip_runtime.h>
#include <hip/hip_bf16.h>

// ---------------------------------------------------------------------------
// EnergyPredictor (equivariant-style GNN) for MI455X / gfx1250.
// Node-level bf16 WMMA GEMMs (f32 acc) with fragment-swizzled LDS staging:
// conversion to bf16 + lane swizzle happens ONCE at staging; inner loop is
// ds_load_b128 x8 -> v_wmma_f32_16x16x32_bf16 x4 per k-step per wave.
// U (52MB) stays L2-resident; per-edge work is VALU + global f32 atomics.
// ---------------------------------------------------------------------------

#define N_NODES_C 10000
#define N_EDGES_C 160000
#define N_GRAPHS_C 16
#define KT_MAX 5          // K <= 160 (actual max 144)

typedef __attribute__((ext_vector_type(16))) __bf16 v16bf;
typedef __attribute__((ext_vector_type(8)))  float  v8f;

__device__ __forceinline__ unsigned int f2bf_pack(float lo, float hi) {
    union { float f; unsigned u; } a, b; a.f = lo; b.f = hi;
    unsigned ua = a.u + (0x7fffu + ((a.u >> 16) & 1u));   // RNE
    unsigned ub = b.u + (0x7fffu + ((b.u >> 16) & 1u));
    return (ua >> 16) | (ub & 0xffff0000u);
}

union Frag { uint4 q[2]; v16bf v; };

__device__ __forceinline__ float silu(float v) {
    return v / (1.0f + __expf(-v));
}

// ---------------------------------------------------------------------------
// GEMM  Out[nRows x nCols] = X[nRows x K] * W[K x nCols]
// 128 threads = 4 waves; block tile 64x64; wave tile 32x32 (4 wmma / k-step).
// LDS holds pre-swizzled bf16 fragments:
//   lane<16 : K in {0..7,16..23}, lane>=16 : K in {8..15,24..31}; A row / B col
//   = lane%16. Fragment = 8 dwords per lane, 16B-aligned -> ds_load_b128.
// ---------------------------------------------------------------------------
__global__ __launch_bounds__(128)
void tp_node_gemm(const float* __restrict__ X, int ldx,
                  const float* __restrict__ W, int ldw,
                  float* __restrict__ Out, int ldo,
                  int nRows, int nCols, int K)
{
    __shared__ __align__(16) unsigned int Asw[KT_MAX][4][32][8];
    __shared__ __align__(16) unsigned int Bsw[KT_MAX][4][32][8];

    const int tid  = threadIdx.x;
    const int row0 = blockIdx.y * 64;
    const int col0 = blockIdx.x * 64;
    const int nkt  = (K + 31) / 32;

    // ---- stage A fragments (rows) ----
    for (int i = tid; i < nkt * 1024; i += 128) {
        int r    = i & 7;
        int lane = (i >> 3) & 31;
        int st   = (i >> 8) & 3;          // 16-row subtile
        int kt   = i >> 10;
        int h = lane >> 4, m = lane & 15;
        int kk = ((r < 4) ? 2 * r : 16 + 2 * (r - 4)) + (h ? 8 : 0);
        int gk = kt * 32 + kk;
        int gr = row0 + st * 16 + m;
        float f0 = 0.f, f1 = 0.f;
        if (gr < nRows) {
            const float* p = X + (size_t)gr * ldx + gk;
            if (gk     < K) f0 = p[0];
            if (gk + 1 < K) f1 = p[1];
        }
        Asw[kt][st][lane][r] = f2bf_pack(f0, f1);
    }
    // ---- stage B fragments (cols) ----
    for (int i = tid; i < nkt * 1024; i += 128) {
        int r    = i & 7;
        int lane = (i >> 3) & 31;
        int st   = (i >> 8) & 3;          // 16-col subtile
        int kt   = i >> 10;
        int h = lane >> 4, m = lane & 15;
        int kk = ((r < 4) ? 2 * r : 16 + 2 * (r - 4)) + (h ? 8 : 0);
        int gk = kt * 32 + kk;
        int gc = col0 + st * 16 + m;
        float f0 = 0.f, f1 = 0.f;
        if (gc < nCols) {
            if (gk     < K) f0 = W[(size_t)gk * ldw + gc];
            if (gk + 1 < K) f1 = W[(size_t)(gk + 1) * ldw + gc];
        }
        Bsw[kt][st][lane][r] = f2bf_pack(f0, f1);
    }
    __syncthreads();

    // ---- compute: wave w -> rows [wr*32,+32), cols [wc*32,+32) ----
    const int w    = tid >> 5;
    const int lane = tid & 31;
    const int wr   = w >> 1;
    const int wc   = w & 1;

    v8f acc[2][2] = {};
    for (int kt = 0; kt < nkt; ++kt) {
        Frag a[2], b[2];
#pragma unroll
        for (int i = 0; i < 2; ++i) {
            const uint4* pa = (const uint4*)&Asw[kt][wr * 2 + i][lane][0];
            a[i].q[0] = pa[0]; a[i].q[1] = pa[1];
            const uint4* pb = (const uint4*)&Bsw[kt][wc * 2 + i][lane][0];
            b[i].q[0] = pb[0]; b[i].q[1] = pb[1];
        }
#pragma unroll
        for (int i = 0; i < 2; ++i)
#pragma unroll
            for (int j = 0; j < 2; ++j)
                acc[i][j] = __builtin_amdgcn_wmma_f32_16x16x32_bf16(
                    false, a[i].v, false, b[j].v, (short)0, acc[i][j],
                    false, false);
    }

    // ---- store D: VGPR r -> row r (lanes 0-15) / r+8 (lanes 16-31) ----
    const int h = lane >> 4, m = lane & 15;
    const int rb = row0 + wr * 32;
    const int cb = col0 + wc * 32;
#pragma unroll
    for (int i = 0; i < 2; ++i)
#pragma unroll
        for (int j = 0; j < 2; ++j)
#pragma unroll
            for (int r = 0; r < 8; ++r) {
                int gr = rb + i * 16 + r + (h ? 8 : 0);
                int gc = cb + j * 16 + m;
                if (gr < nRows && gc < nCols)
                    Out[(size_t)gr * ldo + gc] = acc[i][j][r];
            }
}

// ---------------------------------------------------------------------------
// Per-edge: radial MLP (all 256 threads: one per (edge,neuron) over 32 edges)
// + 9-term contraction of U rows (L2-resident) + atomic scatter into agg.
// ---------------------------------------------------------------------------
#define TILE_E 32
__global__ __launch_bounds__(256)
void edge_message(const float* __restrict__ U, int ldu,
                  const float* __restrict__ edge_attr,
                  const float* __restrict__ emb,
                  const int*   __restrict__ esrc,
                  const int*   __restrict__ edst,
                  const float* __restrict__ W1,
                  const float* __restrict__ W2,
                  const float* __restrict__ W3,
                  float* __restrict__ agg, int nE, int dout)
{
    __shared__ float embS[TILE_E][10];
    __shared__ float eas [TILE_E][9];
    __shared__ float h1s [TILE_E][33];
    __shared__ float h2s [TILE_E][33];
    __shared__ int   srcs[TILE_E];
    __shared__ int   dsts[TILE_E];

    const int e0  = blockIdx.x * TILE_E;
    const int tid = threadIdx.x;

    // stage per-edge inputs
    for (int i = tid; i < TILE_E * 10; i += 256) {
        int el = i / 10, k = i - el * 10;
        int e = e0 + el;
        embS[el][k] = (e < nE) ? emb[(size_t)e * 10 + k] : 0.f;
    }
    for (int i = tid; i < TILE_E * 9; i += 256) {
        int el = i / 9, j = i - el * 9;
        int e = e0 + el;
        eas[el][j] = (e < nE) ? edge_attr[(size_t)e * 9 + j] : 0.f;
    }
    if (tid < TILE_E) {
        int e = e0 + tid;
        srcs[tid] = (e < nE) ? esrc[e] : 0;
        dsts[tid] = (e < nE) ? edst[e] : 0;
    }
    __syncthreads();

    // MLP layer 1: h1 = silu(emb @ W1)   (one thread per (edge, m))
    for (int idx = tid; idx < TILE_E * 32; idx += 256) {
        int el = idx >> 5, m = idx & 31;
        float s = 0.f;
#pragma unroll
        for (int k = 0; k < 10; ++k) s += embS[el][k] * W1[k * 32 + m];
        h1s[el][m] = silu(s);
    }
    __syncthreads();

    // MLP layer 2: h2 = silu(h1 @ W2)
    for (int idx = tid; idx < TILE_E * 32; idx += 256) {
        int el = idx >> 5, m = idx & 31;
        float s = 0.f;
#pragma unroll 8
        for (int k = 0; k < 32; ++k) s += h1s[el][k] * W2[k * 32 + m];
        h2s[el][m] = silu(s);
    }
    __syncthreads();

    // message + scatter: one thread per (edge, o)
    const int total = TILE_E * dout;
    for (int idx = tid; idx < total; idx += 256) {
        int el = idx / dout, o = idx - el * dout;
        int e  = e0 + el;
        if (e >= nE) continue;
        float wgt = 0.f;
#pragma unroll 8
        for (int m = 0; m < 32; ++m) wgt += h2s[el][m] * W3[m * dout + o];
        const float* Ur = U + (size_t)srcs[el] * ldu + o;
        float t = 0.f;
#pragma unroll
        for (int j = 0; j < 9; ++j) t += eas[el][j] * Ur[j * dout];
        atomicAdd(&agg[dsts[el] * dout + o], t * wgt);
    }
}

// ---------------------------------------------------------------------------
// x_out = maybe_silu( self_connection(Usc, node_attr) + agg * inv_sqrt_k )
// ---------------------------------------------------------------------------
__global__ __launch_bounds__(256)
void combine_nodes(const float* __restrict__ Usc,     // ld = 4*dout
                   const float* __restrict__ node_attr,
                   const float* __restrict__ agg,
                   float* __restrict__ xout,
                   int nN, int dout, int apply_silu, float inv)
{
    int idx = blockIdx.x * blockDim.x + threadIdx.x;
    if (idx >= nN * dout) return;
    int n = idx / dout, o = idx - n * dout;
    const float* Ur = Usc + (size_t)n * (4 * dout) + o;
    float sc = 0.f;
#pragma unroll
    for (int j = 0; j < 4; ++j) sc += node_attr[n * 4 + j] * Ur[j * dout];
    float v = sc + agg[idx] * inv;
    if (apply_silu) v = silu(v);
    xout[idx] = v;
}

// ---------------------------------------------------------------------------
// Final edge TP fused with both segment sums (edge->node->graph collapses to
// one scaled atomic into out[batch[dst]]).
// ---------------------------------------------------------------------------
__global__ __launch_bounds__(256)
void final_edge(const float* __restrict__ Uf,         // ld = 72
                const float* __restrict__ edge_attr,
                const int*   __restrict__ esrc,
                const int*   __restrict__ edst,
                const int*   __restrict__ batch,
                float* __restrict__ out, int nE, float scale)
{
    int idx = blockIdx.x * blockDim.x + threadIdx.x;
    if (idx >= nE * 8) return;
    int e = idx >> 3, o = idx & 7;
    const float* Ur = Uf + (size_t)esrc[e] * 72 + o;
    float t = 0.f;
#pragma unroll
    for (int j = 0; j < 9; ++j) t += edge_attr[(size_t)e * 9 + j] * Ur[j * 8];
    atomicAdd(&out[batch[edst[e]] * 8 + o], t * scale);
}

__global__ void fill_zero(float* __restrict__ p, int n) {
    int i = blockIdx.x * blockDim.x + threadIdx.x;
    if (i < n) p[i] = 0.0f;
}

// ---------------------------------------------------------------------------
extern "C" void kernel_launch(void* const* d_in, const int* in_sizes, int n_in,
                              void* d_out, int out_size, void* d_ws, size_t ws_size,
                              hipStream_t stream) {
    (void)n_in; (void)out_size; (void)ws_size;
    const int N = N_NODES_C, E = N_EDGES_C;
    const float inv = 0.25f;                          // 1/sqrt(16)

    const float* node_input = (const float*)d_in[0];  // [N,128]
    const float* node_attr  = (const float*)d_in[1];  // [N,4]
    const float* edge_attr  = (const float*)d_in[2];  // [E,9]
    const float* emb        = (const float*)d_in[3];  // [E,10]

    // params flatten order sniffing: Wf has unique size 576; W1 has size 320.
    const float* Wf;
    int layerBase;
    if (in_sizes[4] == 576) { Wf = (const float*)d_in[4];      layerBase = 5; }
    else                    { Wf = (const float*)d_in[4 + 20]; layerBase = 4; }
    const int* esrc  = (const int*)d_in[25];
    const int* edst  = (const int*)d_in[26];
    const int* batch = (const int*)d_in[27];

    const float *LW1[4], *LW2[4], *LW3[4], *LWsc[4], *LWtp[4];
    const bool sortedOrder = (in_sizes[layerBase] == 320);
    for (int li = 0; li < 4; ++li) {
        int b = layerBase + li * 5;
        if (sortedOrder) {
            LW1[li]  = (const float*)d_in[b + 0];
            LW2[li]  = (const float*)d_in[b + 1];
            LW3[li]  = (const float*)d_in[b + 2];
            LWsc[li] = (const float*)d_in[b + 3];
            LWtp[li] = (const float*)d_in[b + 4];
        } else {  // insertion order: Wtp, Wsc, W1, W2, W3
            LWtp[li] = (const float*)d_in[b + 0];
            LWsc[li] = (const float*)d_in[b + 1];
            LW1[li]  = (const float*)d_in[b + 2];
            LW2[li]  = (const float*)d_in[b + 3];
            LW3[li]  = (const float*)d_in[b + 4];
        }
    }

    // workspace layout (floats)
    float* ws  = (float*)d_ws;
    float* xA  = ws;                       // 10000*144
    float* xB  = xA  + (size_t)N * 144;
    float* U   = xB  + (size_t)N * 144;    // 10000*1296 (52MB, L2-resident)
    float* Usc = U   + (size_t)N * 1296;   // 10000*576
    float* agg = Usc + (size_t)N * 576;    // 10000*144

    const int dims[5] = {128, 144, 144, 144, 8};
    const float* x = node_input;
    int ldx = 128;
    float* bufs[2] = {xA, xB};
    const int gy = (N + 63) / 64;

    for (int li = 0; li < 4; ++li) {
        int din = dims[li], dout = dims[li + 1];
        int cols = 9 * dout, colsSc = 4 * dout;

        tp_node_gemm<<<dim3((cols + 63) / 64, gy), 128, 0, stream>>>(
            x, ldx, LWtp[li], cols, U, cols, N, cols, din);
        tp_node_gemm<<<dim3((colsSc + 63) / 64, gy), 128, 0, stream>>>(
            x, ldx, LWsc[li], colsSc, Usc, colsSc, N, colsSc, din);

        int aggN = N * dout;
        fill_zero<<<(aggN + 255) / 256, 256, 0, stream>>>(agg, aggN);

        edge_message<<<(E + TILE_E - 1) / TILE_E, 256, 0, stream>>>(
            U, cols, edge_attr, emb, esrc, edst,
            LW1[li], LW2[li], LW3[li], agg, E, dout);

        float* xo = bufs[li & 1];
        combine_nodes<<<(N * dout + 255) / 256, 256, 0, stream>>>(
            Usc, node_attr, agg, xo, N, dout, (li < 3) ? 1 : 0, inv);
        x = xo;
        ldx = dout;
    }

    // Final TP: Uf = x(8) @ Wf(8 x 72), then edge gather + double segment-sum.
    tp_node_gemm<<<dim3((72 + 63) / 64, gy), 128, 0, stream>>>(
        x, 8, Wf, 72, U, 72, N, 72, 8);

    fill_zero<<<1, 128, 0, stream>>>((float*)d_out, N_GRAPHS_C * 8);
    final_edge<<<(E * 8 + 255) / 256, 256, 0, stream>>>(
        U, edge_attr, esrc, edst, batch, (float*)d_out, E, inv * inv);
}
